// FactoredLSTM_79353815761156
// MI455X (gfx1250) — matrix operand order repaired and checked by reading, not compiled, not run on hardware
//
#include <hip/hip_runtime.h>

typedef __attribute__((ext_vector_type(16))) __bf16 v16bf;
typedef __attribute__((ext_vector_type(8)))  __bf16 v8bf;
typedef __attribute__((ext_vector_type(8)))  float  v8f;

#define B_    32
#define L_    64
#define T_    63      // L-1 timesteps
#define EMB_  512
#define STY_  512
#define STY4_ 2048
#define HID_  1024
#define HID4_ 4096
#define VOC_  32000
#define AST   520     // padded LDS row stride (bf16) for 512-wide tiles
#define AST2  264     // padded LDS row stride (bf16) for 256-wide tiles

// ---------------------------------------------------------------- fragments
static __device__ __forceinline__ v16bf make_frag(v8bf lo, v8bf hi) {
  v16bf r;
#pragma unroll
  for (int i = 0; i < 8; ++i) { r[i] = lo[i]; r[i + 8] = hi[i]; }
  return r;
}

// A fragment 16x32 bf16, row-major source. Per CDNA5 ISA: lanes 0-15 hold
// row M=lane, K {0..7} in v0-3 and K {16..23} in v4-7; lanes 16-31 hold
// K {8..15} and {24..31}. base = &A[row0*stride + k0].
static __device__ __forceinline__ v16bf load_a_frag(const __bf16* base, int stride, int lane) {
  const int half = lane >> 4, lr = lane & 15;
  const __bf16* p = base + (size_t)lr * stride + half * 8;
  v8bf lo = *(const v8bf*)p;
  v8bf hi = *(const v8bf*)(p + 16);
  return make_frag(lo, hi);
}

// B fragment 32x16 bf16 from BT[N][K] (transposed weight). Same K layout as
// A: lanes 0-15 (col N=lane) hold K {0..7} and {16..23}, lanes
// 16-31 hold K {8..15} and {24..31}. base = &BT[n0*K + k0].
static __device__ __forceinline__ v16bf load_b_frag(const __bf16* base, int strideK, int lane) {
  const int half = lane >> 4, lr = lane & 15;
  const __bf16* p = base + (size_t)lr * strideK + half * 8;
  v8bf lo = *(const v8bf*)p;
  v8bf hi = *(const v8bf*)(p + 16);
  return make_frag(lo, hi);
}

static __device__ __forceinline__ float sigmoidf_(float x) {
  return 1.0f / (1.0f + __expf(-x));
}

#define WMMA_BF16(a, b, c) \
  __builtin_amdgcn_wmma_f32_16x16x32_bf16(false, (a), false, (b), (short)0, (c), false, false)

// ---------------------------------------------------------------- generic GEMM
// C[M,N] = A[M,K](bf16, row-major) * BT[N,K](bf16) + bias[N]
// out_mode: 0 = f32 row-major [M,N]; 1 = bf16 row-major [M,N]
__global__ __launch_bounds__(256) void gemm_bf16_wmma(
    const __bf16* __restrict__ A, const __bf16* __restrict__ BT,
    const float* __restrict__ bias, void* __restrict__ out,
    int M, int N, int K, int out_mode)
{
  __shared__ __bf16 Alds[32 * AST];
  const int tid  = threadIdx.x;
  const int wave = tid >> 5, lane = tid & 31;
  const int half = lane >> 4, lr = lane & 15;
  const int m0 = blockIdx.y * 32;
  const int n0 = blockIdx.x * 256 + wave * 32;

  v8f acc[2][2];
#pragma unroll
  for (int a = 0; a < 2; ++a)
#pragma unroll
    for (int b = 0; b < 2; ++b)
#pragma unroll
      for (int i = 0; i < 8; ++i) acc[a][b][i] = 0.0f;

  for (int kc = 0; kc < K; kc += 512) {
    for (int ch = tid; ch < 32 * 64; ch += 256) {
      int r = ch >> 6, cc = (ch & 63) << 3;
      *(v8bf*)&Alds[r * AST + cc] = *(const v8bf*)&A[(size_t)(m0 + r) * K + kc + cc];
    }
    __syncthreads();
#pragma unroll 4
    for (int k = 0; k < 512; k += 32) {
      v16bf a0 = load_a_frag(&Alds[0  * AST + k], AST, lane);
      v16bf a1 = load_a_frag(&Alds[16 * AST + k], AST, lane);
      v16bf b0 = load_b_frag(&BT[(size_t)(n0 +  0) * K + kc + k], K, lane);
      v16bf b1 = load_b_frag(&BT[(size_t)(n0 + 16) * K + kc + k], K, lane);
      acc[0][0] = WMMA_BF16(a0, b0, acc[0][0]);
      acc[0][1] = WMMA_BF16(a0, b1, acc[0][1]);
      acc[1][0] = WMMA_BF16(a1, b0, acc[1][0]);
      acc[1][1] = WMMA_BF16(a1, b1, acc[1][1]);
    }
    __syncthreads();
  }

#pragma unroll
  for (int mt = 0; mt < 2; ++mt) {
#pragma unroll
    for (int nt = 0; nt < 2; ++nt) {
      const int col = n0 + nt * 16 + lr;
      const float bv = bias[col];
#pragma unroll
      for (int i = 0; i < 8; ++i) {
        const int row = m0 + mt * 16 + half * 8 + i;
        const float v = acc[mt][nt][i] + bv;
        if (out_mode == 0) ((float*)out)[(size_t)row * N + col] = v;
        else               ((__bf16*)out)[(size_t)row * N + col] = (__bf16)v;
      }
    }
  }
}

// ---------------------------------------------------------------- output GEMM
// Specialized dominant GEMM: A = Hall [2048 x 1024] bf16 (rows 0..31 are h0,
// skipped at store), BT = PwT [32000 x 1024], bias = P_b. WG tile 64x256,
// wave: 4 Mtiles x 2 Ntiles -> 8 WMMA per k-step against 2 B frags (halves
// L2 traffic on the 65MB bf16 P matrix vs the generic kernel).
// Scatter store: row m -> t = m/32 - 1, b = m%32; out[b][t][col].
__global__ __launch_bounds__(256) void gemm_out_wmma(
    const __bf16* __restrict__ A, const __bf16* __restrict__ BT,
    const float* __restrict__ bias, float* __restrict__ out)
{
  __shared__ __bf16 Alds[64 * AST2];
  const int tid  = threadIdx.x;
  const int wave = tid >> 5, lane = tid & 31;
  const int half = lane >> 4, lr = lane & 15;
  const int m0 = blockIdx.y * 64;
  const int n0 = blockIdx.x * 256 + wave * 32;
  const int K  = HID_;

  v8f acc[4][2];
#pragma unroll
  for (int a = 0; a < 4; ++a)
#pragma unroll
    for (int b = 0; b < 2; ++b)
#pragma unroll
      for (int i = 0; i < 8; ++i) acc[a][b][i] = 0.0f;

  for (int kc = 0; kc < K; kc += 256) {
    // stage A chunk: 64 rows x 256 bf16
    for (int ch = tid; ch < 64 * 32; ch += 256) {
      int r = ch >> 5, cc = (ch & 31) << 3;
      *(v8bf*)&Alds[r * AST2 + cc] = *(const v8bf*)&A[(size_t)(m0 + r) * K + kc + cc];
    }
    __syncthreads();
#pragma unroll 2
    for (int k = 0; k < 256; k += 32) {
      v16bf b0 = load_b_frag(&BT[(size_t)(n0 +  0) * K + kc + k], K, lane);
      v16bf b1 = load_b_frag(&BT[(size_t)(n0 + 16) * K + kc + k], K, lane);
#pragma unroll
      for (int mt = 0; mt < 4; ++mt) {
        v16bf a = load_a_frag(&Alds[(mt * 16) * AST2 + k], AST2, lane);
        acc[mt][0] = WMMA_BF16(a, b0, acc[mt][0]);
        acc[mt][1] = WMMA_BF16(a, b1, acc[mt][1]);
      }
    }
    __syncthreads();
  }

#pragma unroll
  for (int mt = 0; mt < 4; ++mt) {
#pragma unroll
    for (int nt = 0; nt < 2; ++nt) {
      const int col = n0 + nt * 16 + lr;
      const float bv = bias[col];
#pragma unroll
      for (int i = 0; i < 8; ++i) {
        const int row = m0 + mt * 16 + half * 8 + i;
        const int tt = (row >> 5) - 1;   // row 0..31 = h0, not an output step
        const int bb = row & 31;
        if (tt >= 0)
          out[(size_t)bb * ((size_t)T_ * VOC_) + (size_t)tt * VOC_ + col] =
              acc[mt][nt][i] + bv;
      }
    }
  }
}

// ---------------------------------------------------------------- LSTM step
__global__ __launch_bounds__(256) void lstm_step_kernel(
    const __bf16* __restrict__ Hprev, __bf16* __restrict__ Hnext,
    const __bf16* __restrict__ WT, const float* __restrict__ Gx,
    const float* __restrict__ Wb, float* __restrict__ Cst)
{
  __shared__ float g_lds[4][32][65];
  const int tid  = threadIdx.x;
  const int wave = tid >> 5, lane = tid & 31;
  const int half = lane >> 4, lr = lane & 15;
  const int wg   = blockIdx.x;
  const int gate = wave >> 1;
  const int ncl  = (wave & 1) * 32;
  const int gcol = gate * HID_ + wg * 64 + ncl;

  v8f acc[2][2];
#pragma unroll
  for (int a = 0; a < 2; ++a)
#pragma unroll
    for (int b = 0; b < 2; ++b)
#pragma unroll
      for (int i = 0; i < 8; ++i) acc[a][b][i] = 0.0f;

#pragma unroll 4
  for (int k = 0; k < HID_; k += 32) {
    v16bf a0 = load_a_frag(&Hprev[0  * HID_ + k], HID_, lane);
    v16bf a1 = load_a_frag(&Hprev[16 * HID_ + k], HID_, lane);
    v16bf b0 = load_b_frag(&WT[(size_t)(gcol +  0) * HID_ + k], HID_, lane);
    v16bf b1 = load_b_frag(&WT[(size_t)(gcol + 16) * HID_ + k], HID_, lane);
    acc[0][0] = WMMA_BF16(a0, b0, acc[0][0]);
    acc[0][1] = WMMA_BF16(a0, b1, acc[0][1]);
    acc[1][0] = WMMA_BF16(a1, b0, acc[1][0]);
    acc[1][1] = WMMA_BF16(a1, b1, acc[1][1]);
  }

#pragma unroll
  for (int mt = 0; mt < 2; ++mt)
#pragma unroll
    for (int nt = 0; nt < 2; ++nt)
#pragma unroll
      for (int i = 0; i < 8; ++i)
        g_lds[gate][mt * 16 + half * 8 + i][ncl + nt * 16 + lr] = acc[mt][nt][i];
  __syncthreads();

  for (int idx = tid; idx < 32 * 64; idx += 256) {
    const int row = idx >> 6, col = idx & 63;
    const int j = wg * 64 + col;
    const float* gx = Gx + (size_t)row * HID4_;
    const float iv = g_lds[0][row][col] + gx[j]             + Wb[j];
    const float fv = g_lds[1][row][col] + gx[HID_ + j]      + Wb[HID_ + j];
    const float ov = g_lds[2][row][col] + gx[2 * HID_ + j]  + Wb[2 * HID_ + j];
    const float cg = g_lds[3][row][col] + gx[3 * HID_ + j]  + Wb[3 * HID_ + j];
    const float cold = Cst[row * HID_ + j];
    const float cnew = sigmoidf_(fv) * cold + sigmoidf_(iv) * tanhf(cg);
    const float hnew = sigmoidf_(ov) * cnew;
    Cst[row * HID_ + j] = cnew;
    Hnext[row * HID_ + j] = (__bf16)hnew;
  }
}

// ---------------------------------------------------------------- prep kernels
__global__ void transpose_f32_to_bf16(const float* __restrict__ in,
                                      __bf16* __restrict__ out, int R, int C) {
  size_t i = (size_t)blockIdx.x * blockDim.x + threadIdx.x;
  if (i >= (size_t)R * C) return;
  int r = (int)(i / C), c = (int)(i % C);
  out[(size_t)c * R + r] = (__bf16)in[i];
}

__global__ void gather_embed_kernel(const int* __restrict__ tok,
                                    const float* __restrict__ embed,
                                    __bf16* __restrict__ X) {
  size_t i = (size_t)blockIdx.x * blockDim.x + threadIdx.x;
  if (i >= (size_t)T_ * B_ * EMB_) return;
  int e = (int)(i % EMB_);
  int m = (int)(i / EMB_);
  int t = m >> 5, b = m & 31;           // row m = t*32 + b, t in [0,62]
  int token = tok[b * L_ + t];          // x_t = emb[:, t]
  X[i] = (__bf16)embed[(size_t)token * EMB_ + e];
}

__global__ void init_hc_kernel(const float* __restrict__ sv,
                               __bf16* __restrict__ H0, float* __restrict__ Cst) {
  int i = blockIdx.x * blockDim.x + threadIdx.x;
  if (i < B_ * HID_) { H0[i] = (__bf16)sv[i]; Cst[i] = sv[i]; }
}

// ---------------------------------------------------------------- launch
extern "C" void kernel_launch(void* const* d_in, const int* in_sizes, int n_in,
                              void* d_out, int out_size, void* d_ws, size_t ws_size,
                              hipStream_t stream) {
  const int*   tok   = (const int*)d_in[0];
  const float* sv    = (const float*)d_in[1];
  const float* embed = (const float*)d_in[2];
  const float* U_w   = (const float*)d_in[3];
  const float* U_b   = (const float*)d_in[4];
  const float* S1_w  = (const float*)d_in[5];
  const float* S1_b  = (const float*)d_in[6];
  const float* V_w   = (const float*)d_in[7];
  const float* V_b   = (const float*)d_in[8];
  const float* W_w   = (const float*)d_in[9];
  const float* W_b   = (const float*)d_in[10];
  const float* P_w   = (const float*)d_in[11];
  const float* P_b   = (const float*)d_in[12];
  (void)in_sizes; (void)n_in; (void)out_size; (void)ws_size;

  char* ws = (char*)d_ws;
  size_t off = 0;
  auto alloc = [&](size_t bytes) -> void* {
    void* p = ws + off;
    off += (bytes + 255) & ~(size_t)255;
    return p;
  };
  const size_t M = (size_t)T_ * B_;  // 2016
  __bf16* Xbf  = (__bf16*)alloc(M * EMB_ * 2);
  __bf16* UwT  = (__bf16*)alloc((size_t)STY4_ * EMB_ * 2);
  __bf16* S1wT = (__bf16*)alloc((size_t)STY_  * STY4_ * 2);
  __bf16* VwT  = (__bf16*)alloc((size_t)HID4_ * STY_ * 2);
  __bf16* WwT  = (__bf16*)alloc((size_t)HID4_ * HID_ * 2);
  __bf16* PwT  = (__bf16*)alloc((size_t)VOC_  * HID_ * 2);
  __bf16* Ubf  = (__bf16*)alloc(M * STY4_ * 2);
  __bf16* Sbf  = (__bf16*)alloc(M * STY_ * 2);
  float*  Gx   = (float*) alloc(M * HID4_ * 4);
  __bf16* Hall = (__bf16*)alloc((size_t)(T_ + 1) * B_ * HID_ * 2); // 2048 rows
  float*  Cst  = (float*) alloc((size_t)B_ * HID_ * 4);

  auto cdiv = [](size_t a, size_t b) { return (unsigned)((a + b - 1) / b); };

  // 1) weight convert + transpose to bf16 [N][K]
  transpose_f32_to_bf16<<<cdiv((size_t)EMB_ * STY4_, 256), 256, 0, stream>>>(U_w,  UwT,  EMB_,  STY4_);
  transpose_f32_to_bf16<<<cdiv((size_t)STY4_ * STY_, 256), 256, 0, stream>>>(S1_w, S1wT, STY4_, STY_);
  transpose_f32_to_bf16<<<cdiv((size_t)STY_ * HID4_, 256), 256, 0, stream>>>(V_w,  VwT,  STY_,  HID4_);
  transpose_f32_to_bf16<<<cdiv((size_t)HID_ * HID4_, 256), 256, 0, stream>>>(W_w,  WwT,  HID_,  HID4_);
  transpose_f32_to_bf16<<<cdiv((size_t)HID_ * VOC_, 256), 256, 0, stream>>>(P_w,  PwT,  HID_,  VOC_);

  // 2) embedding gather + state init
  gather_embed_kernel<<<cdiv(M * EMB_, 256), 256, 0, stream>>>(tok, embed, Xbf);
  init_hc_kernel<<<cdiv((size_t)B_ * HID_, 256), 256, 0, stream>>>(sv, Hall, Cst);

  // 3) hoisted input-side GEMMs over all 2016 rows
  gemm_bf16_wmma<<<dim3(STY4_ / 256, (unsigned)(M / 32)), 256, 0, stream>>>(
      Xbf, UwT, U_b, Ubf, (int)M, STY4_, EMB_, 1);
  gemm_bf16_wmma<<<dim3(STY_ / 256, (unsigned)(M / 32)), 256, 0, stream>>>(
      Ubf, S1wT, S1_b, Sbf, (int)M, STY_, STY4_, 1);
  gemm_bf16_wmma<<<dim3(HID4_ / 256, (unsigned)(M / 32)), 256, 0, stream>>>(
      Sbf, VwT, V_b, Gx, (int)M, HID4_, STY_, 0);

  // 4) sequential recurrence (stream order = dependency)
  for (int t = 0; t < T_; ++t) {
    lstm_step_kernel<<<16, 256, 0, stream>>>(
        Hall + (size_t)t * B_ * HID_,
        Hall + (size_t)(t + 1) * B_ * HID_,
        WwT,
        Gx + (size_t)t * B_ * HID4_,
        W_b, Cst);
  }

  // 5) output projection over all 2048 Hall rows (h0 rows skipped at store)
  gemm_out_wmma<<<dim3(VOC_ / 256, (unsigned)((T_ + 1) * B_ / 64)), 256, 0, stream>>>(
      Hall, PwT, P_b, (float*)d_out);
}
